// GetDensity_49435073577717
// MI455X (gfx1250) — compile-verified, matrix operand order checked by reading, not run on hardware
//
#include <hip/hip_runtime.h>

// Problem constants (match reference)
#define NLOCAL  10000
#define NW      8        // nwave
#define NORB    128      // norbit
#define NANG    13       // 1 + 3 + 9 angular channels
#define CHAN    (NANG*NW)  // 104 floats per atom accumulator row
#define PI_OVER_CUTOFF 0.6283185307179586f  // pi / 5.0

typedef __attribute__((ext_vector_type(2)))  float    v2f;
typedef __attribute__((ext_vector_type(8)))  float    v8f;
typedef __attribute__((ext_vector_type(16))) _Float16 v16h;

// ---------------------------------------------------------------------------
// Kernel 1: acc[i,j,w] = ef_ang[j] * ef_para[w]   (folds ef_orbital into the
// segment-sum accumulator so the scatter kernel just atomically adds on top)
// ---------------------------------------------------------------------------
__global__ void init_acc_kernel(const float* __restrict__ ef,
                                const float* __restrict__ ef_para,
                                float* __restrict__ acc, int total) {
    int idx = blockIdx.x * blockDim.x + threadIdx.x;
    if (idx >= total) return;
    int rem = idx % CHAN;
    int j = rem >> 3;          // angular channel 0..12
    int w = rem & 7;           // wave 0..7
    float e;
    if (j == 0)      e = 1.0f;
    else if (j < 4)  e = ef[j - 1];
    else {
        int jj = (j - 4) / 3, kk = (j - 4) % 3;
        e = ef[jj] * ef[kk];
    }
    acc[idx] = e * ef_para[w];
}

// ---------------------------------------------------------------------------
// Kernel 2: per-pair radial*angular features, scattered via L2 f32 atomics.
// acc is 4.16 MB -> resident in the 192MB L2 where the atomics execute.
// ---------------------------------------------------------------------------
__global__ void pair_scatter_kernel(const float* __restrict__ cart,
                                    const int*   __restrict__ atom_index,
                                    const int*   __restrict__ local_species,
                                    const int*   __restrict__ neigh_list,
                                    const float* __restrict__ rs,
                                    const float* __restrict__ inta,
                                    const float* __restrict__ params,
                                    float* __restrict__ acc, int M) {
    int m = blockIdx.x * blockDim.x + threadIdx.x;
    if (m >= M) return;

    const int i0 = atom_index[m];        // atom_index[0][m]
    const int i1 = atom_index[M + m];    // atom_index[1][m]

    const float dx = cart[i0 * 3 + 0] - cart[i1 * 3 + 0];
    const float dy = cart[i0 * 3 + 1] - cart[i1 * 3 + 1];
    const float dz = cart[i0 * 3 + 2] - cart[i1 * 3 + 2];
    const float d  = sqrtf(dx * dx + dy * dy + dz * dz);

    float fc = 0.5f * __cosf(d * PI_OVER_CUTOFF) + 0.5f;
    fc *= fc;

    const int t = local_species[neigh_list[m]];

    // radial * orb_coeff (params uses the same species index)
    float wrad[NW];
#pragma unroll
    for (int w = 0; w < NW; ++w) {
        const float dr = d - rs[t * NW + w];
        wrad[w] = __expf(inta[t * NW + w] * dr * dr) * params[t * NW + w];
    }

    float ang[NANG];
    const float dv[3] = {dx, dy, dz};
    ang[0] = fc;
#pragma unroll
    for (int a = 0; a < 3; ++a) ang[1 + a] = fc * dv[a];
#pragma unroll
    for (int a = 0; a < 3; ++a)
#pragma unroll
        for (int b = 0; b < 3; ++b) ang[4 + a * 3 + b] = ang[1 + a] * dv[b];

    float* dst = acc + (size_t)i0 * CHAN;
#pragma unroll
    for (int j = 0; j < NANG; ++j) {
#pragma unroll
        for (int w = 0; w < NW; ++w) {
            atomicAdd(dst + j * NW + w, ang[j] * wrad[w]);
        }
    }
}

// ---------------------------------------------------------------------------
// Kernel 3: per-channel GEMM [16 atoms x 8] x [8 x 16 orbitals] via
// V_WMMA_F32_16X16X4_F32 (2 K-steps cover K=8, exact f32), square fragment,
// reduce over 13 channels, store density tile.
// Block = 256 threads = 8 waves; wave w owns orbit tile w (8*16 = 128 = NORB).
// Grid = 625 blocks * 16 atoms = exactly NLOCAL -> EXEC all-1s at every WMMA.
// ---------------------------------------------------------------------------
__global__ __launch_bounds__(256)
void density_wmma_kernel(const float* __restrict__ acc,
                         const float* __restrict__ hyper,
                         float* __restrict__ out) {
    const int lane      = threadIdx.x & 31;
    const int wave      = threadIdx.x >> 5;
    const int atom_base = blockIdx.x * 16;
    const int o_base    = wave * 16;
    const int half      = lane >> 4;   // 0: lanes 0-15, 1: lanes 16-31
    const int mn        = lane & 15;   // A: row M; B/C: col N

    float dacc[8] = {0.f, 0.f, 0.f, 0.f, 0.f, 0.f, 0.f, 0.f};

#pragma unroll
    for (int j = 0; j < NANG; ++j) {
        const int p = (j == 0) ? 0 : ((j < 4) ? 1 : 2);   // INDEX_PARA[j]
        const float* abase = acc + ((size_t)(atom_base + mn) * NANG + j) * NW;
        const float* bbase = hyper + (size_t)p * NW * NORB + o_base + mn;

        v8f c = {};
#if __has_builtin(__builtin_amdgcn_wmma_f32_16x16x4_f32)
        // A 16x4 f32: lane half h -> VGPR r holds K = q*4 + 2h + r (contiguous pair)
        // B 4x16 f32 mirrored: VGPR r holds row K = q*4 + 2h + r at col N=mn
#pragma unroll
        for (int q = 0; q < 2; ++q) {
            const int k0 = q * 4 + half * 2;
            v2f a = *reinterpret_cast<const v2f*>(abase + k0);
            v2f b;
            b.x = bbase[(size_t)(k0 + 0) * NORB];
            b.y = bbase[(size_t)(k0 + 1) * NORB];
            c = __builtin_amdgcn_wmma_f32_16x16x4_f32(
                    false, a, false, b, (short)0, c, false, false);
        }
#else
        // Fallback: codegen-confirmed f16 K=32 WMMA, K=8 used, rest zero.
        // 16-bit A layout: lanes 0-15 VGPR0-3 carry K0..7 packed pairs.
        v16h a16, b16;
#pragma unroll
        for (int e = 0; e < 16; ++e) { a16[e] = (_Float16)0.f; b16[e] = (_Float16)0.f; }
        if (half == 0) {
#pragma unroll
            for (int k = 0; k < NW; ++k) {
                a16[k] = (_Float16)abase[k];
                b16[k] = (_Float16)bbase[(size_t)k * NORB];
            }
        }
        c = __builtin_amdgcn_wmma_f32_16x16x32_f16(
                false, a16, false, b16, (short)0, c, false, false);
#endif
#pragma unroll
        for (int r = 0; r < 8; ++r) dacc[r] += c[r] * c[r];
    }

    // C/D layout: VGPR r, lane l -> M = r + 8*(l/16), N = l%16
#pragma unroll
    for (int r = 0; r < 8; ++r) {
        const int atom = atom_base + r + 8 * half;
        out[(size_t)atom * NORB + o_base + mn] = dacc[r];
    }
}

// ---------------------------------------------------------------------------
extern "C" void kernel_launch(void* const* d_in, const int* in_sizes, int n_in,
                              void* d_out, int out_size, void* d_ws, size_t ws_size,
                              hipStream_t stream) {
    const float* cart          = (const float*)d_in[0];
    const float* ef            = (const float*)d_in[1];
    const int*   atom_index    = (const int*)  d_in[2];
    const int*   local_species = (const int*)  d_in[3];
    const int*   neigh_list    = (const int*)  d_in[4];
    const float* rs            = (const float*)d_in[5];
    const float* inta          = (const float*)d_in[6];
    const float* params        = (const float*)d_in[7];
    const float* hyper         = (const float*)d_in[8];
    const float* ef_para       = (const float*)d_in[9];

    float* out = (float*)d_out;
    float* acc = (float*)d_ws;          // NLOCAL*13*8 floats = 4.16 MB scratch

    const int nlocal = in_sizes[3];     // 10000
    const int M      = in_sizes[4];     // 400000
    const int total  = nlocal * CHAN;

    hipLaunchKernelGGL(init_acc_kernel,
                       dim3((total + 255) / 256), dim3(256), 0, stream,
                       ef, ef_para, acc, total);

    hipLaunchKernelGGL(pair_scatter_kernel,
                       dim3((M + 255) / 256), dim3(256), 0, stream,
                       cart, atom_index, local_species, neigh_list,
                       rs, inta, params, acc, M);

    hipLaunchKernelGGL(density_wmma_kernel,
                       dim3(nlocal / 16), dim3(256), 0, stream,
                       acc, hyper, out);
}
